// SMMILe_51634096833149
// MI455X (gfx1250) — matrix operand order repaired and checked by reading, not compiled, not run on hardware
//
#include <hip/hip_runtime.h>

typedef unsigned short u16;
typedef unsigned int   u32;
typedef unsigned long long u64;

typedef __attribute__((ext_vector_type(16))) __bf16 bf16x16;
typedef __attribute__((ext_vector_type(8)))  float  f32x8;
typedef __attribute__((ext_vector_type(2)))  float  f32x2;
typedef __attribute__((ext_vector_type(4)))  u32    u32x4;
typedef __attribute__((ext_vector_type(8)))  int    i32x8;
typedef __attribute__((ext_vector_type(4)))  int    i32x4;

#define KPATCH 65536
#define HGRID  256
#define DIN    512
#define DOUT   256
#define NCLS   5
#define NSP    2048
#define NDROP  1024
#define TILE_CS 520   /* 512 channels + 8 u16 pad -> kills LDS bank conflicts */

__device__ __forceinline__ u16 f2bf(float f) {
  u32 b = __float_as_uint(f);
  u32 r = b + 0x7FFFu + ((b >> 16) & 1u);   // round-to-nearest-even
  return (u16)(r >> 16);
}

// ---------------------------------------------------------------- casts/prep
__global__ void k_cast_x(const float* __restrict__ x, u16* __restrict__ xb) {
  int i = blockIdx.x * 256 + threadIdx.x;          // one element each
  xb[i] = f2bf(x[i]);
}

// conv_w HWIO (3,3,512,256) f32  ->  Wb[tap][co][ci] bf16 (k-contiguous B rows)
__global__ void k_prep_w(const float* __restrict__ cw, u16* __restrict__ Wb) {
  int i = blockIdx.x * 256 + threadIdx.x;          // 9*512*256 threads
  int t  = i / (DIN * DOUT);
  int r  = i % (DIN * DOUT);
  int ci = r / DOUT;
  int co = r % DOUT;
  Wb[((size_t)t * DOUT + co) * DIN + ci] = f2bf(cw[i]);
}

// ---------------------------------------------------------------- TDM helper
// 1D contiguous bf16 copy global->LDS with LDS padding: 4 dwords pad after
// every 256 dwords (one 512-channel column block) -> column stride 520 u16.
__device__ __forceinline__ void tdm_load_row(u32 lds_addr, const void* gp, u32 nelem) {
  u64 ga = (u64)(uintptr_t)gp;
  u32x4 g0;
  g0[0] = 1u;                                   // count=1, user descriptor
  g0[1] = lds_addr;                             // lds_addr [63:32]
  g0[2] = (u32)ga;                              // global_addr lo
  g0[3] = (u32)(ga >> 32) | (2u << 30);         // global_addr hi | type=2
  i32x8 g1;
  g1[0] = (int)((1u << 16)                      // data_size = 2 bytes
              | (1u << 20)                      // pad_enable
              | (7u << 22)                      // pad_interval: 256 dwords
              | (3u << 25));                    // pad_amount: 4 dwords (16B)
  g1[1] = (int)((nelem & 0xFFFFu) << 16);       // tensor_dim0 lo16 @ bits 63:48
  g1[2] = (int)((nelem >> 16) | (1u << 16));    // tensor_dim0 hi16 | tensor_dim1=1
  g1[3] = (int)(nelem << 16);                   // tensor_dim1 hi=0 | tile_dim0
  g1[4] = 1;                                    // tile_dim1=1, tile_dim2=0
  g1[5] = (int)nelem;                           // tensor_dim0_stride lo32
  g1[6] = 0;
  g1[7] = 0;
  i32x4 gz = (i32x4)0;
#if __clang_major__ >= 23
  __builtin_amdgcn_tensor_load_to_lds(g0, g1, gz, gz, (i32x8)0, 0);
#else
  __builtin_amdgcn_tensor_load_to_lds(g0, g1, gz, gz, 0);
#endif
}

// ---------------------------------------------------------------- conv (WMMA)
// block = 256 thr = 8 waves; computes 16 patches (one row segment) x 256 ch.
__global__ __launch_bounds__(256) void k_conv(const u16* __restrict__ xb,
                                              const u16* __restrict__ Wb,
                                              const float* __restrict__ conv_b,
                                              float* __restrict__ h) {
  __shared__ u16 tile[3 * 18 * TILE_CS];        // 56,160 B
  const int wg = blockIdx.x;
  const int y  = wg >> 4;
  const int x0 = (wg & 15) << 4;
  const int t    = threadIdx.x;
  const int lane = t & 31;
  const int wv   = t >> 5;

  // zero-fill halo regions that the DMA won't cover (grid borders)
  for (int r = 0; r < 3; ++r) {
    int yy = y + r - 1;
    if (yy < 0 || yy >= HGRID)
      for (int i = t; i < 18 * TILE_CS; i += 256) tile[r * 18 * TILE_CS + i] = 0;
  }
  if (x0 == 0)
    for (int r = 0; r < 3; ++r)
      for (int i = t; i < DIN; i += 256) tile[r * 18 * TILE_CS + i] = 0;
  if (x0 == HGRID - 16)
    for (int r = 0; r < 3; ++r)
      for (int i = t; i < DIN; i += 256) tile[(r * 18 + 17) * TILE_CS + i] = 0;

  // wave 0 drives the Tensor Data Mover: one 1D row-copy per input row
  if (wv == 0) {
    for (int r = 0; r < 3; ++r) {
      int yy = y + r - 1;
      if (yy < 0 || yy >= HGRID) continue;
      int c0 = x0 - 1, dst = 0, ncols = 18;
      if (c0 < 0)            { c0 = 0; dst = 1; ncols = 17; }
      if (x0 == HGRID - 16)  { ncols = 17; }
      u32 lds_addr = (u32)(uintptr_t)&tile[(r * 18 + dst) * TILE_CS];
      const u16* gsrc = xb + ((size_t)yy * HGRID + c0) * DIN;
      tdm_load_row(lds_addr, gsrc, (u32)(ncols * DIN));
    }
    __builtin_amdgcn_s_wait_tensorcnt(0);
  }
  __syncthreads();

  const int m = lane & 15, half = lane >> 4;
  const int n0 = wv * 2;                        // two 16-ch N tiles per wave
  union Frag { bf16x16 v; u32x4 q[2]; } A, B0, B1;
  f32x8 acc0 = (f32x8)0.0f, acc1 = (f32x8)0.0f;

  #pragma unroll
  for (int tap = 0; tap < 9; ++tap) {
    const int dy = tap / 3, dx = tap % 3;
    const u16* wt    = Wb + (size_t)tap * DOUT * DIN;
    const u16* arow  = &tile[((dy * 18) + (m + dx)) * TILE_CS + half * 16];
    const u16* b0row = wt + (size_t)(n0 * 16 + m) * DIN + half * 16;
    const u16* b1row = b0row + 16 * DIN;
    for (int c = 0; c < 16; ++c) {              // 512 ch / 32 per WMMA
      const u16* ap = arow + c * 32;
      A.q[0] = *(const u32x4*)(ap);
      A.q[1] = *(const u32x4*)(ap + 8);
      const u16* bp0 = b0row + c * 32;
      B0.q[0] = *(const u32x4*)(bp0);
      B0.q[1] = *(const u32x4*)(bp0 + 8);
      const u16* bp1 = b1row + c * 32;
      B1.q[0] = *(const u32x4*)(bp1);
      B1.q[1] = *(const u32x4*)(bp1 + 8);
      acc0 = __builtin_amdgcn_wmma_f32_16x16x32_bf16(false, A.v, false, B0.v,
                                                     (short)0, acc0, false, false);
      acc1 = __builtin_amdgcn_wmma_f32_16x16x32_bf16(false, A.v, false, B1.v,
                                                     (short)0, acc1, false, false);
    }
  }

  const int base_p = y * HGRID + x0;
  const int ch0 = n0 * 16 + m, ch1 = ch0 + 16;
  const float b0 = conv_b[ch0], b1 = conv_b[ch1];
  #pragma unroll
  for (int r = 0; r < 8; ++r) {
    int p = base_p + r + half * 8;
    h[(size_t)p * DOUT + ch0] = fmaxf(acc0[r] + b0, 0.0f);
    h[(size_t)p * DOUT + ch1] = fmaxf(acc1[r] + b1, 0.0f);
  }
}

// ---------------------------------------------------------------- inst scores
__global__ void k_inst(const float* __restrict__ h, const float* __restrict__ cls_w,
                       const float* __restrict__ cls_b, float* __restrict__ inst) {
  int lane = threadIdx.x & 31, wv = threadIdx.x >> 5;
  int p = blockIdx.x * 8 + wv;
  float acc[NCLS] = {0, 0, 0, 0, 0};
  for (int d = lane; d < DOUT; d += 32) {
    float hv = h[(size_t)p * DOUT + d];
    #pragma unroll
    for (int j = 0; j < NCLS; ++j) acc[j] += hv * cls_w[j * DOUT + d];
  }
  #pragma unroll
  for (int j = 0; j < NCLS; ++j) {
    float v = acc[j];
    for (int off = 16; off; off >>= 1) v += __shfl_down(v, off, 32);
    if (lane == 0) inst[(size_t)p * NCLS + j] = v + cls_b[j];
  }
}

// ---------------------------------------------------------------- sampling
__global__ void k_seginit(u64* segkey) { segkey[blockIdx.x * 256 + threadIdx.x] = 0ull; }

__global__ void k_sample(const int* __restrict__ sp, u64* __restrict__ segkey) {
  int k = blockIdx.x * 256 + threadIdx.x;
  u32 x = (u32)k * 2654435761u;
  x ^= x >> 16; x *= 2246822519u; x ^= x >> 13; x *= 3266489917u; x ^= x >> 16;
  float r = (float)(x >> 8) * (1.0f / 16777216.0f);          // uniform surrogate
  u64 key = ((u64)__float_as_uint(r) << 32) | (u32)k;        // argmax-in-segment
  atomicMax(&segkey[sp[k]], key);
}

__global__ void k_gather(const u64* __restrict__ segkey, const float* __restrict__ h,
                         float* __restrict__ hb) {
  int s = blockIdx.x, d = threadIdx.x;
  int idx = (int)(segkey[s] & 0xFFFFFFFFull);
  hb[(size_t)s * DOUT + d] = h[(size_t)idx * DOUT + d];
}

// ---------------------------------------------------------------- gated attn (f32 WMMA)
__global__ __launch_bounds__(256) void k_attn(const float* __restrict__ hb,
                                              const float* __restrict__ Vw,
                                              const float* __restrict__ Vb,
                                              const float* __restrict__ Uw,
                                              const float* __restrict__ Ub,
                                              float* __restrict__ g) {
  int lane = threadIdx.x & 31, wv = threadIdx.x >> 5;
  int tid = blockIdx.x * 8 + wv;         // 2048 wave tiles
  int mt = tid >> 4, nt = tid & 15;
  int m = lane & 15, half = lane >> 4;
  f32x8 va = (f32x8)0.0f, ua = (f32x8)0.0f;
  for (int kc = 0; kc < DOUT; kc += 4) {
    int k0 = kc + half * 2;
    f32x2 a, bv, bu;
    const float* hrow = hb + (size_t)(mt * 16 + m) * DOUT + k0;
    a.x = hrow[0]; a.y = hrow[1];
    int col = nt * 16 + m;                          // lane&15 doubles as N col
    bv.x = Vw[(size_t)k0 * DOUT + col]; bv.y = Vw[(size_t)(k0 + 1) * DOUT + col];
    bu.x = Uw[(size_t)k0 * DOUT + col]; bu.y = Uw[(size_t)(k0 + 1) * DOUT + col];
    va = __builtin_amdgcn_wmma_f32_16x16x4_f32(false, a, false, bv, (short)0, va, false, false);
    ua = __builtin_amdgcn_wmma_f32_16x16x4_f32(false, a, false, bu, (short)0, ua, false, false);
  }
  int col = nt * 16 + m;
  float vb = Vb[col], ub = Ub[col];
  #pragma unroll
  for (int r = 0; r < 8; ++r) {
    int row = mt * 16 + r + half * 8;
    float tv = tanhf(va[r] + vb);
    float su = 1.0f / (1.0f + expf(-(ua[r] + ub)));
    g[(size_t)row * DOUT + col] = tv * su;
  }
}

__global__ void k_ascore(const float* __restrict__ g, const float* __restrict__ aw,
                         const float* __restrict__ ab, float* __restrict__ Araw) {
  int lane = threadIdx.x & 31, wv = threadIdx.x >> 5;
  int s = blockIdx.x * 8 + wv;
  float acc[NCLS] = {0, 0, 0, 0, 0};
  for (int d = lane; d < DOUT; d += 32) {
    float gv = g[(size_t)s * DOUT + d];
    #pragma unroll
    for (int j = 0; j < NCLS; ++j) acc[j] += gv * aw[d * NCLS + j];
  }
  #pragma unroll
  for (int j = 0; j < NCLS; ++j) {
    float v = acc[j];
    for (int off = 16; off; off >>= 1) v += __shfl_down(v, off, 32);
    if (lane == 0) Araw[j * NSP + s] = v + ab[j];
  }
}

// ---------------------------------------------------------------- softmax / InD
__global__ void k_softmax(const float* __restrict__ Araw, float* __restrict__ A) {
  int c = blockIdx.x, t = threadIdx.x;
  __shared__ float red[256];
  const float* row = Araw + c * NSP;
  float mx = -3.0e38f;
  for (int i = t; i < NSP; i += 256) mx = fmaxf(mx, row[i]);
  red[t] = mx; __syncthreads();
  for (int off = 128; off; off >>= 1) { if (t < off) red[t] = fmaxf(red[t], red[t + off]); __syncthreads(); }
  mx = red[0]; __syncthreads();
  float s = 0.f;
  for (int i = t; i < NSP; i += 256) s += expf(row[i] - mx);
  red[t] = s; __syncthreads();
  for (int off = 128; off; off >>= 1) { if (t < off) red[t] += red[t + off]; __syncthreads(); }
  float inv = 1.0f / red[0];
  for (int i = t; i < NSP; i += 256) A[c * NSP + i] = expf(row[i] - mx) * inv;
}

__global__ void k_ind(const float* __restrict__ Araw, float* __restrict__ Adrop) {
  int c = blockIdx.x, t = threadIdx.x;
  __shared__ u32 u[NSP];
  __shared__ int icnt;
  __shared__ u32 take;
  __shared__ float red[256];
  const float* row = Araw + c * NSP;
  for (int i = t; i < NSP; i += 256) {
    u32 b = __float_as_uint(row[i]);
    u[i] = (b & 0x80000000u) ? ~b : (b | 0x80000000u);   // order-preserving map
  }
  __syncthreads();
  u32 prefix = 0;                                        // radix-select rank NDROP
  for (int bit = 31; bit >= 0; --bit) {
    u32 cand = prefix | (1u << bit);
    if (t == 0) icnt = 0;
    __syncthreads();
    int local = 0;
    for (int i = t; i < NSP; i += 256) local += (u[i] >= cand) ? 1 : 0;
    atomicAdd(&icnt, local);
    __syncthreads();
    if (icnt >= NDROP) prefix = cand;
    __syncthreads();
  }
  if (t == 0) { icnt = 0; take = 0; }
  __syncthreads();
  { int local = 0;
    for (int i = t; i < NSP; i += 256) local += (u[i] > prefix) ? 1 : 0;
    atomicAdd(&icnt, local); }
  __syncthreads();
  int need_eq = NDROP - icnt;
  for (int i = t; i < NSP; i += 256) {
    u32 key = u[i], drop = 0;
    if (key > prefix) drop = 1;
    else if (key == prefix) { u32 o = atomicAdd(&take, 1u); if ((int)o < need_eq) drop = 1; }
    u[i] = drop;
  }
  __syncthreads();
  float mx = -3.0e38f;
  for (int i = t; i < NSP; i += 256) if (!u[i]) mx = fmaxf(mx, row[i]);
  red[t] = mx; __syncthreads();
  for (int off = 128; off; off >>= 1) { if (t < off) red[t] = fmaxf(red[t], red[t + off]); __syncthreads(); }
  mx = red[0]; __syncthreads();
  float s = 0.f;
  for (int i = t; i < NSP; i += 256) if (!u[i]) s += expf(row[i] - mx);
  red[t] = s; __syncthreads();
  for (int off = 128; off; off >>= 1) { if (t < off) red[t] += red[t + off]; __syncthreads(); }
  float inv = 1.0f / red[0];
  for (int i = t; i < NSP; i += 256) Adrop[c * NSP + i] = u[i] ? 0.f : expf(row[i] - mx) * inv;
}

__global__ void k_bag(const float* __restrict__ Adrop, const float* __restrict__ hb,
                      const float* __restrict__ cls_w, const float* __restrict__ cls_b,
                      float* __restrict__ bag) {
  int c = blockIdx.x, d = threadIdx.x;
  float m = 0.f;
  for (int s = 0; s < NSP; ++s) m += Adrop[c * NSP + s] * hb[(size_t)s * DOUT + d];
  float v = m * cls_w[c * DOUT + d];
  __shared__ float red[256];
  red[d] = v; __syncthreads();
  for (int off = 128; off; off >>= 1) { if (d < off) red[d] += red[d + off]; __syncthreads(); }
  if (d == 0) bag[c] = red[0] + cls_b[c];
}

// ---------------------------------------------------------------- launch
extern "C" void kernel_launch(void* const* d_in, const int* in_sizes, int n_in,
                              void* d_out, int out_size, void* d_ws, size_t ws_size,
                              hipStream_t stream) {
  const float* x      = (const float*)d_in[0];
  const int*   sp     = (const int*)  d_in[1];
  const float* conv_w = (const float*)d_in[2];
  const float* conv_b = (const float*)d_in[3];
  const float* V_w    = (const float*)d_in[4];
  const float* V_b    = (const float*)d_in[5];
  const float* U_w    = (const float*)d_in[6];
  const float* U_b    = (const float*)d_in[7];
  const float* attn_w = (const float*)d_in[8];
  const float* attn_b = (const float*)d_in[9];
  const float* cls_w  = (const float*)d_in[10];
  const float* cls_b  = (const float*)d_in[11];

  float* out   = (float*)d_out;
  float* bag   = out;                         // 5
  float* Araw  = out + 5;                     // 5*2048
  float* A     = out + 5 + NCLS * NSP;        // 5*2048
  float* h     = out + 5 + 2 * NCLS * NSP;    // 65536*256
  float* inst  = h + (size_t)KPATCH * DOUT;   // 65536*5

  char* ws = (char*)d_ws;
  u16*  xb     = (u16*)ws;                                      // 64 MiB
  u16*  Wb     = (u16*)(ws + (size_t)67108864);                 // 2.25 MiB
  float* hb    = (float*)(ws + (size_t)69468160);               // 2 MiB
  float* g     = (float*)(ws + (size_t)71565312);               // 2 MiB
  u64*  segkey = (u64*)  (ws + (size_t)73662464);               // 16 KiB
  float* Adrop = (float*)(ws + (size_t)73678848);               // 40 KiB

  k_cast_x  <<<KPATCH * DIN / 256, 256, 0, stream>>>(x, xb);
  k_prep_w  <<<9 * DIN * DOUT / 256, 256, 0, stream>>>(conv_w, Wb);
  k_conv    <<<KPATCH / 16, 256, 0, stream>>>(xb, Wb, conv_b, h);
  k_inst    <<<KPATCH / 8, 256, 0, stream>>>(h, cls_w, cls_b, inst);
  k_seginit <<<NSP / 256, 256, 0, stream>>>(segkey);
  k_sample  <<<KPATCH / 256, 256, 0, stream>>>(sp, segkey);
  k_gather  <<<NSP, 256, 0, stream>>>(segkey, h, hb);
  k_attn    <<<NSP / 8, 256, 0, stream>>>(hb, V_w, V_b, U_w, U_b, g);
  k_ascore  <<<NSP / 8, 256, 0, stream>>>(g, attn_w, attn_b, Araw);
  k_softmax <<<NCLS, 256, 0, stream>>>(Araw, A);
  k_ind     <<<NCLS, 256, 0, stream>>>(Araw, Adrop);
  k_bag     <<<NCLS, 256, 0, stream>>>(Adrop, hb, cls_w, cls_b, bag);
}
